// qkv_proj_19722489823947
// MI455X (gfx1250) — matrix operand order repaired and checked by reading, not compiled, not run on hardware
//
#include <hip/hip_runtime.h>

// ---------------------------------------------------------------------------
// QKV projection for MI455X (gfx1250, wave32, WMMA).
//   C[8192,3072] = X[8192,1024] * W^T + b, scattered to q/k/v [4,16,2048,64].
// Precision: bf16x3 split (hi/lo) with f32 WMMA accumulation (~fp32 accuracy,
// ~2.7x the native f32 16x16x4 WMMA throughput).
// Round 2: hardware v_cvt_pk_bf16_f32 conversions + LDS double buffering
// (1 barrier per k-step, ping/pong resolved at compile time via x2 unroll).
// ---------------------------------------------------------------------------

typedef __attribute__((ext_vector_type(16))) __bf16        v16bf;
typedef __attribute__((ext_vector_type(4)))  __bf16        v4bf;
typedef __attribute__((ext_vector_type(8)))  float         v8f;
typedef __attribute__((ext_vector_type(4)))  float         v4f;
typedef __attribute__((ext_vector_type(8)))  unsigned short v8u16;
typedef __attribute__((ext_vector_type(16))) unsigned short v16u16;

namespace {
constexpr int kB = 4, kS = 2048, kDin = 1024, kDout = 1024, kH = 16;
constexpr int kM = kB * kS;       // 8192 rows (B*S)
constexpr int kN = 3 * kDout;     // 3072 cols
constexpr int kK = kDin;          // 1024
constexpr int BM = 128, BN = 128, BK = 32;
constexpr int LDK  = 40;          // padded u16 row stride (80B: 16B-aligned frags, conflict-free)
constexpr int TILE = BM * LDK;    // 5120 u16 per LDS array
constexpr int BUF  = 4 * TILE;    // Ah|Al|Bh|Bl per k-step buffer
constexpr int KSTEPS = kK / BK;   // 32
constexpr int HD = kDout / kH;    // 64 head dim
}

__device__ __forceinline__ v16u16 join8(v8u16 a, v8u16 b) {
  return __builtin_shufflevector(a, b, 0,1,2,3,4,5,6,7,8,9,10,11,12,13,14,15);
}

__global__ __launch_bounds__(256) void qkv_wmma_bf16x3(
    const float* __restrict__ X, const float* __restrict__ W,
    const float* __restrict__ bias, float* __restrict__ out) {
  __shared__ __align__(32) unsigned short smem[2 * BUF];  // 80 KB, double buffered

  const int tid  = threadIdx.x;
  const int bm   = blockIdx.y, bn = blockIdx.x;
  const int wave = tid >> 5, lane = tid & 31;
  const int wm   = wave >> 2, wn = wave & 3;      // 2 (M) x 4 (N) waves
  const int lrow = lane & 15, lhi = lane >> 4;

  const float* Ag = X + (size_t)bm * BM * kK;
  const float* Bg = W + (size_t)bn * BN * kK;     // W row n == column n of B

  // ---- global tile load: 128x32 f32 each for A and B (float4 per thread x4)
  auto g_load = [&](int kk, v4f (&a)[4], v4f (&b)[4]) {
#pragma unroll
    for (int i = 0; i < 4; ++i) {
      int idx = i * 256 + tid;
      int row = idx >> 3, kq = idx & 7;
      a[i] = *(const v4f*)(Ag + (size_t)row * kK + kk + kq * 4);
      b[i] = *(const v4f*)(Bg + (size_t)row * kK + kk + kq * 4);
    }
  };

  // ---- split to bf16 hi/lo (hardware RNE cvt) and stage in LDS
  auto lds_store = [&](unsigned short* dst, const v4f (&a)[4], const v4f (&b)[4]) {
    unsigned short* Ah = dst;
    unsigned short* Al = dst + TILE;
    unsigned short* Bh = dst + 2 * TILE;
    unsigned short* Bl = dst + 3 * TILE;
#pragma unroll
    for (int i = 0; i < 4; ++i) {
      int idx = i * 256 + tid;
      int row = idx >> 3, kq = idx & 7;
      int off = row * LDK + kq * 4;
      v4bf ah, al, bh, bl;
#pragma unroll
      for (int j = 0; j < 4; ++j) {
        __bf16 h  = (__bf16)a[i][j];                 // v_cvt_pk_bf16_f32 (RNE)
        al[j]     = (__bf16)(a[i][j] - (float)h);    // residual -> lo
        ah[j]     = h;
        __bf16 hb = (__bf16)b[i][j];
        bl[j]     = (__bf16)(b[i][j] - (float)hb);
        bh[j]     = hb;
      }
      *(v4bf*)(Ah + off) = ah;
      *(v4bf*)(Al + off) = al;
      *(v4bf*)(Bh + off) = bh;
      *(v4bf*)(Bl + off) = bl;
    }
  };

  // ---- one K-step: load fragments in contiguous k order (same for A and B), 24 WMMAs
  auto k_step = [&](v8f (&acc)[4][2], const unsigned short* src) {
    const unsigned short* Ah = src;
    const unsigned short* Al = src + TILE;
    const unsigned short* Bh = src + 2 * TILE;
    const unsigned short* Bl = src + 3 * TILE;
    v16u16 afh[4], afl[4];
#pragma unroll
    for (int mt = 0; mt < 4; ++mt) {
      // A 16x32 bf16: M=lane&15, K = 16*(lane>>4) + 0..15 contiguous
      int r = (wm * 64 + mt * 16 + lrow) * LDK + lhi * 16;
      afh[mt] = join8(*(const v8u16*)(Ah + r), *(const v8u16*)(Ah + r + 8));
      afl[mt] = join8(*(const v8u16*)(Al + r), *(const v8u16*)(Al + r + 8));
    }
#pragma unroll
    for (int nt = 0; nt < 2; ++nt) {
      // B 32x16 bf16: N=lane&15, K = 16*(lane>>4) + 0..15 contiguous
      int r = (wn * 32 + nt * 16 + lrow) * LDK + lhi * 16;
      v16bf bh = __builtin_bit_cast(v16bf, join8(*(const v8u16*)(Bh + r), *(const v8u16*)(Bh + r + 8)));
      v16bf bl = __builtin_bit_cast(v16bf, join8(*(const v8u16*)(Bl + r), *(const v8u16*)(Bl + r + 8)));
#pragma unroll
      for (int mt = 0; mt < 4; ++mt) {
        v16bf ah = __builtin_bit_cast(v16bf, afh[mt]);
        v16bf al = __builtin_bit_cast(v16bf, afl[mt]);
        acc[mt][nt] = __builtin_amdgcn_wmma_f32_16x16x32_bf16(false, ah, false, bh, (short)0, acc[mt][nt], false, false);
        acc[mt][nt] = __builtin_amdgcn_wmma_f32_16x16x32_bf16(false, ah, false, bl, (short)0, acc[mt][nt], false, false);
        acc[mt][nt] = __builtin_amdgcn_wmma_f32_16x16x32_bf16(false, al, false, bh, (short)0, acc[mt][nt], false, false);
      }
    }
  };

  v8f acc[4][2];
  v8f z = {0.f, 0.f, 0.f, 0.f, 0.f, 0.f, 0.f, 0.f};
#pragma unroll
  for (int mt = 0; mt < 4; ++mt)
#pragma unroll
    for (int nt = 0; nt < 2; ++nt) acc[mt][nt] = z;

  unsigned short* ping = smem;
  unsigned short* pong = smem + BUF;

  {
    v4f a0[4], b0[4];
    g_load(0, a0, b0);
    lds_store(ping, a0, b0);
  }
  __syncthreads();

  // One k-step: read `cur`, prefetch k+1 to regs, stage into `nxt`, 1 barrier.
  auto body = [&](int ks, unsigned short* cur, unsigned short* nxt) {
    v4f aN[4], bN[4];
    const bool more = (ks + 1) < KSTEPS;
    if (more) g_load((ks + 1) * BK, aN, bN);   // global loads in flight over WMMAs
    k_step(acc, cur);                          // frag reads consumed pre-barrier
    if (more) lds_store(nxt, aN, bN);          // writes go to the other buffer
    __syncthreads();
  };

  for (int ks = 0; ks < KSTEPS; ks += 2) {     // x2: ping/pong known statically
    body(ks,     ping, pong);
    body(ks + 1, pong, ping);
  }

  // ---- epilogue: bias + scatter into q/k/v [B,H,S,64]
#pragma unroll
  for (int nt = 0; nt < 2; ++nt) {
    int n = bn * BN + wn * 32 + nt * 16 + lrow;  // global column
    float bv = bias[n];
    int h     = n / (3 * HD);
    int r     = n - h * (3 * HD);
    int which = r >> 6;          // 0=q 1=k 2=v
    int d     = r & (HD - 1);
    float* obase = out + (size_t)which * ((size_t)kB * kH * kS * HD);
#pragma unroll
    for (int mt = 0; mt < 4; ++mt) {
#pragma unroll
      for (int j = 0; j < 8; ++j) {
        int row = bm * BM + wm * 64 + mt * 16 + lhi * 8 + j;  // C layout: VGPR j -> M=j / j+8
        int bi = row >> 11;
        int s  = row & (kS - 1);
        size_t off = ((size_t)((bi * kH + h) * kS + s) << 6) + d;
        obase[off] = acc[mt][nt][j] + bv;
      }
    }
  }
}

extern "C" void kernel_launch(void* const* d_in, const int* in_sizes, int n_in,
                              void* d_out, int out_size, void* d_ws, size_t ws_size,
                              hipStream_t stream) {
  (void)in_sizes; (void)n_in; (void)d_ws; (void)ws_size; (void)out_size;
  const float* X    = (const float*)d_in[0];
  const float* W    = (const float*)d_in[1];
  const float* bias = (const float*)d_in[2];
  float* out        = (float*)d_out;
  dim3 grid(kN / BN, kM / BM);  // (24, 64) = 1536 workgroups
  qkv_wmma_bf16x3<<<grid, dim3(256, 1, 1), 0, stream>>>(X, W, bias, out);
}
